// integratedAutoencoder_20675972563481
// MI455X (gfx1250) — matrix operand order repaired and checked by reading, not compiled, not run on hardware
//
#include <hip/hip_runtime.h>
#include <hip/hip_bf16.h>
#include <math.h>

// CDNA5 (gfx1250) fused 10-net autoencoder forward.
// Prep kernels pre-convert x / weights into fragment-major f16 blobs in d_ws;
// the main kernel streams them into LDS with gfx1250 async-to-LDS copies and
// runs wave32 v_wmma_f32_16x16x32_f16 on contiguous 32B fragments.

typedef __attribute__((ext_vector_type(16))) _Float16 v16h;
typedef __attribute__((ext_vector_type(8)))  float    v8f;
typedef __attribute__((ext_vector_type(4)))  unsigned int v4u;
typedef __attribute__((ext_vector_type(4)))  int v4i;

#define BATCH    8192
#define NNETS    10
#define BT       128         // batch rows per workgroup
#define NTHREADS 256         // 8 wave32s

#if __has_builtin(__builtin_amdgcn_global_load_async_to_lds_b128) && \
    __has_builtin(__builtin_amdgcn_s_wait_asynccnt)
#define USE_ASYNC 1
#else
#define USE_ASYNC 0
#endif

#if __has_builtin(__builtin_amdgcn_sched_barrier)
#define SCHED_FENCE() __builtin_amdgcn_sched_barrier(0)
#else
#define SCHED_FENCE()
#endif

// ws layout (halves): XF | W0F | WSF | W7F
#define XF_HALVES   ((size_t)64 * 25 * 4096)   // x frags: [block][kc][rt*32+lane][16]
#define W0F_HALVES  ((size_t)10 * 25 * 4096)   // W0 frags: [net][kc][nt*32+lane][16]
#define WSF_HALVES  ((size_t)10 * 20480)       // W1..W6 frags per net
#define W7F_HALVES  ((size_t)10 * 7 * 14336)   // W7 frags: [net][nc][(nt*4+kc)*32+lane][16]

__device__ __forceinline__ v8f wmma16(v16h a, v16h b, v8f c) {
  return __builtin_amdgcn_wmma_f32_16x16x32_f16(
      false, a, false, b, (short)0, c, false, false);
}

// exact-form fast tanh: 1 - 2/(e^{2x}+1); saturates to +/-1, NaN propagates.
__device__ __forceinline__ float fast_tanh(float x) {
  const float e = __expf(2.f * x);
  return 1.f - 2.f * __builtin_amdgcn_rcpf(e + 1.f);
}

// Scatter an activation value (row,col) into an A-fragment-major f16 buffer.
__device__ __forceinline__ void store_A_elem(_Float16* dst, int DKC,
                                             int row, int col, float v) {
  const int rt   = row >> 4, m = row & 15;
  const int kc   = col >> 5, k = col & 31;
  const int lane = m + ((k & 8) << 1);
  const int h    = (k & 7) + ((k & 16) >> 1);
  dst[(size_t)(((rt * DKC) + kc) * 32 + lane) * 16 + h] = (_Float16)v;
}

// Raw block copy global(f16 frags) -> LDS, 16B per thread-iteration.
__device__ __forceinline__ void copy_lds(_Float16* dst, const _Float16* src,
                                         int halves, int tid) {
#if USE_ASYNC
  for (int i = tid; i < (halves >> 3); i += NTHREADS)
    __builtin_amdgcn_global_load_async_to_lds_b128(
        (__attribute__((address_space(1))) v4i*)(src + (size_t)i * 8),
        (__attribute__((address_space(3))) v4i*)(dst + (size_t)i * 8), 0, 0);
#else
  for (int i = tid; i < (halves >> 3); i += NTHREADS)
    ((v4u*)dst)[i] = ((const v4u*)src)[i];
#endif
}

__device__ __forceinline__ void wait_async() {
#if USE_ASYNC
  __builtin_amdgcn_s_wait_asynccnt(0);
#endif
}

__device__ __forceinline__ void stage_b(const float* bg, int N, float* bl,
                                        int NP, int tid) {
  for (int i = tid; i < NP; i += NTHREADS) bl[i] = (i < N) ? bg[i] : 0.f;
}

// ---------------- prep kernels: fp32 -> fragment-major f16 in d_ws ----------

// x[8192,784] -> per (batch-block, K-chunk) A-frag blobs of 4096 halves.
__global__ __launch_bounds__(256)
void prep_x_kernel(const float* __restrict__ x, _Float16* __restrict__ XF) {
  const int bb = blockIdx.x, kc = blockIdx.y;     // 64 x 25
  const int u = threadIdx.x;                      // 256 dest 32B units
  const int rt = u >> 5, lane = u & 31;
  const int row = bb * BT + rt * 16 + (lane & 15);
  const int kb  = kc * 32 + ((lane & 16) ? 8 : 0);
  _Float16 tmp[16];
  const bool tail = (kc == 24);
#pragma unroll
  for (int h = 0; h < 16; ++h) {
    const int k = kb + ((h < 8) ? h : (h + 8));
    float v;
    if (tail) v = (k < 784) ? x[(size_t)row * 784 + k] : 0.f;
    else      v = x[(size_t)row * 784 + k];
    tmp[h] = (_Float16)v;
  }
  *(v16h*)&XF[((size_t)(bb * 25 + kc) * 256 + u) * 16] = *(v16h*)tmp;
}

// Convert a KxN (leading dim ldn) fp32 weight window into a B-frag blob
// of KC*NT tiles laid out [nt][kc][lane][16h] (zero-padded).
__device__ __forceinline__ void conv_W_frag(const float* __restrict__ Wg,
                                            int K, int ldn, int N,
                                            int KC, int NT,
                                            _Float16* __restrict__ gdst) {
  const int units = KC * NT * 32;
  for (int u = threadIdx.x; u < units; u += 256) {
    const int lane = u & 31, tile = u >> 5;
    const int kc = tile % KC, nt = tile / KC;
    const int n  = nt * 16 + (lane & 15);
    const int kb = kc * 32 + ((lane & 16) ? 8 : 0);
    _Float16 tmp[16];
#pragma unroll
    for (int h = 0; h < 16; ++h) {
      const int k = kb + ((h < 8) ? h : (h + 8));
      tmp[h] = (_Float16)((k < K && n < N) ? Wg[(size_t)k * ldn + n] : 0.f);
    }
    *(v16h*)&gdst[(size_t)u * 16] = *(v16h*)tmp;
  }
}

__global__ __launch_bounds__(256)
void prep_w_kernel(const float* __restrict__ W0, const float* __restrict__ W1,
                   const float* __restrict__ W2, const float* __restrict__ W3,
                   const float* __restrict__ W4, const float* __restrict__ W5,
                   const float* __restrict__ W6, const float* __restrict__ W7,
                   _Float16* __restrict__ W0F, _Float16* __restrict__ WSF,
                   _Float16* __restrict__ W7F) {
  const int net = blockIdx.y, seg = blockIdx.x;   // 38 x 10
  if (seg < 25) {                                 // W0 K-chunk frags (NT=8, KC=1)
    const int kc = seg;
    conv_W_frag(W0 + (size_t)net * 784 * 128 + (size_t)kc * 32 * 128,
                784 - kc * 32, 128, 128, 1, 8,
                W0F + (size_t)(net * 25 + kc) * 4096);
  } else if (seg < 32) {                          // W7 column-chunk frags (KC=4, NT=7)
    const int nc = seg - 25;
    conv_W_frag(W7 + (size_t)net * 128 * 784 + nc * 112,
                128, 784, 112, 4, 7,
                W7F + (size_t)(net * 7 + nc) * 14336);
  } else {                                        // W1..W6 full frags
    const int l = seg - 32;                       // 0..5
    const int Ks[6]   = {128, 64, 12, 3, 12, 64};
    const int Ns[6]   = {64, 12, 3, 12, 64, 128};
    const int KCs[6]  = {4, 2, 1, 1, 1, 2};
    const int NTs[6]  = {4, 1, 1, 1, 4, 8};
    const int offs[6] = {0, 8192, 9216, 9728, 10240, 12288};
    const float* Wp[6] = {W1, W2, W3, W4, W5, W6};
    conv_W_frag(Wp[l] + (size_t)net * Ks[l] * Ns[l], Ks[l], Ns[l], Ns[l],
                KCs[l], NTs[l], WSF + (size_t)net * 20480 + offs[l]);
  }
}

// ---------------- main fused kernel -----------------------------------------

// Fused layer: A frags hoisted, B frags preloaded, sched fence keeps the
// ds_load group ahead of the wmma chain.
template<int KC, int NT, int DKC, int ACT>
__device__ __forceinline__ void layer_small(const _Float16* src, const _Float16* Wf,
                                            const float* bias, _Float16* dst,
                                            int wave, int lane) {
  const v16h* As = (const v16h*)src;
  const v16h* Bs = (const v16h*)Wf;
  const int ncol  = lane & 15;
  const int mbase = wave * 16 + ((lane & 16) ? 8 : 0);
  v16h areg[KC];
#pragma unroll
  for (int kc = 0; kc < KC; ++kc) areg[kc] = As[(wave * KC + kc) * 32 + lane];
#pragma unroll
  for (int nt = 0; nt < NT; ++nt) {
    v16h breg[KC];
#pragma unroll
    for (int kc = 0; kc < KC; ++kc) breg[kc] = Bs[(nt * KC + kc) * 32 + lane];
    SCHED_FENCE();
    v8f acc = {};
#pragma unroll
    for (int kc = 0; kc < KC; ++kc) acc = wmma16(areg[kc], breg[kc], acc);
    const int n = nt * 16 + ncol;
    const float bv = bias[n];
#pragma unroll
    for (int r = 0; r < 8; ++r) {
      float v = acc[r] + bv;
      if (ACT == 0) v = fmaxf(v, 0.f);
      store_A_elem(dst, DKC, mbase + r, n, v);
    }
  }
}

__global__ __launch_bounds__(NTHREADS)
void ae_fused_kernel(const _Float16* __restrict__ XF,
                     const _Float16* __restrict__ W0F,
                     const _Float16* __restrict__ WSF,
                     const _Float16* __restrict__ W7F,
                     const float* __restrict__ b0, const float* __restrict__ b1,
                     const float* __restrict__ b2, const float* __restrict__ b3,
                     const float* __restrict__ b4, const float* __restrict__ b5,
                     const float* __restrict__ b6, const float* __restrict__ b7,
                     float* __restrict__ out) {
  __shared__ alignas(32) _Float16 h1buf[8 * 4 * 512];   // 32 KB L0 out / L6 out
  __shared__ alignas(32) _Float16 hAbuf[8 * 2 * 512];   // 16 KB h2 then h6
  __shared__ alignas(32) _Float16 hBbuf[8 * 1 * 512];   //  8 KB h3 then h5
  __shared__ alignas(32) _Float16 hCbuf[8 * 1 * 512];   //  8 KB h4 (latent)
  __shared__ alignas(32) _Float16 wbuf[16 * 512];       // 16 KB small-layer W frags
  __shared__ alignas(32) _Float16 w7frag[28 * 512];     // 28 KB layer-7 W chunk
  __shared__ alignas(32) _Float16 xfrag[2][8 * 512];    // 16 KB ping-pong
  __shared__ alignas(32) _Float16 w0frag[2][8 * 512];   // 16 KB ping-pong
  __shared__ float biasbuf[128];
  __shared__ float b7buf[112];
  __shared__ float rowsum[BT];

  const int tid  = threadIdx.x;
  const int wave = tid >> 5;
  const int lane = tid & 31;
  const int net  = blockIdx.y;
  const int brow = blockIdx.x * BT;

  for (int i = tid; i < 8 * 512; i += NTHREADS) {
    hBbuf[i] = (_Float16)0.f; hCbuf[i] = (_Float16)0.f;
  }
  if (tid < BT) rowsum[tid] = 0.f;

  // ---------- layer 0: streamed K, async ping-pong ----------
  const _Float16* XFb  = XF  + (size_t)blockIdx.x * 25 * 4096;
  const _Float16* W0Fn = W0F + (size_t)net * 25 * 4096;
  stage_b(b0 + (size_t)net * 128, 128, biasbuf, 128, tid);

  copy_lds(xfrag[0],  XFb,  4096, tid);
  copy_lds(w0frag[0], W0Fn, 4096, tid);

  v8f acc0[8] = {};
  for (int kc = 0; kc < 25; ++kc) {
    wait_async();
    __syncthreads();                              // chunk kc resident
    if (kc < 24) {
      const int nb = (kc + 1) & 1;
      copy_lds(xfrag[nb],  XFb  + (size_t)(kc + 1) * 4096, 4096, tid);
      copy_lds(w0frag[nb], W0Fn + (size_t)(kc + 1) * 4096, 4096, tid);
    }
    const v16h* As = (const v16h*)xfrag[kc & 1];
    const v16h* Bs = (const v16h*)w0frag[kc & 1];
    v16h a = As[wave * 32 + lane];
    v16h bfr[4];
#pragma unroll
    for (int nt = 0; nt < 4; ++nt) bfr[nt] = Bs[nt * 32 + lane];
    SCHED_FENCE();
#pragma unroll
    for (int nt = 0; nt < 4; ++nt) acc0[nt] = wmma16(a, bfr[nt], acc0[nt]);
#pragma unroll
    for (int nt = 0; nt < 4; ++nt) bfr[nt] = Bs[(nt + 4) * 32 + lane];
    SCHED_FENCE();
#pragma unroll
    for (int nt = 0; nt < 4; ++nt) acc0[nt + 4] = wmma16(a, bfr[nt], acc0[nt + 4]);
  }
  {
    const int ncol  = lane & 15;
    const int mbase = wave * 16 + ((lane & 16) ? 8 : 0);
#pragma unroll
    for (int nt = 0; nt < 8; ++nt) {
      const int n = nt * 16 + ncol;
      const float bv = biasbuf[n];
#pragma unroll
      for (int r = 0; r < 8; ++r)
        store_A_elem(h1buf, 4, mbase + r, n, fmaxf(acc0[nt][r] + bv, 0.f));
    }
  }
  __syncthreads();

  // ---------- layers 1..6 ----------
  const _Float16* WSFn = WSF + (size_t)net * 20480;

  copy_lds(wbuf, WSFn + 0, 8192, tid);                       // W1 frags
  stage_b(b1 + (size_t)net * 64, 64, biasbuf, 64, tid);
  wait_async(); __syncthreads();
  layer_small<4, 4, 2, 0>(h1buf, wbuf, biasbuf, hAbuf, wave, lane);   // -> h2
  __syncthreads();

  copy_lds(wbuf, WSFn + 8192, 1024, tid);                    // W2
  stage_b(b2 + (size_t)net * 12, 12, biasbuf, 16, tid);
  wait_async(); __syncthreads();
  layer_small<2, 1, 1, 0>(hAbuf, wbuf, biasbuf, hBbuf, wave, lane);   // -> h3
  __syncthreads();

  copy_lds(wbuf, WSFn + 9216, 512, tid);                     // W3
  stage_b(b3 + (size_t)net * 3, 3, biasbuf, 16, tid);
  wait_async(); __syncthreads();
  layer_small<1, 1, 1, 1>(hBbuf, wbuf, biasbuf, hCbuf, wave, lane);   // -> h4 latent
  __syncthreads();

  copy_lds(wbuf, WSFn + 9728, 512, tid);                     // W4
  stage_b(b4 + (size_t)net * 12, 12, biasbuf, 16, tid);
  wait_async(); __syncthreads();
  layer_small<1, 1, 1, 0>(hCbuf, wbuf, biasbuf, hBbuf, wave, lane);   // -> h5
  __syncthreads();

  copy_lds(wbuf, WSFn + 10240, 2048, tid);                   // W5
  stage_b(b5 + (size_t)net * 64, 64, biasbuf, 64, tid);
  wait_async(); __syncthreads();
  layer_small<1, 4, 2, 0>(hBbuf, wbuf, biasbuf, hAbuf, wave, lane);   // -> h6
  __syncthreads();

  copy_lds(wbuf, WSFn + 12288, 8192, tid);                   // W6
  stage_b(b6 + (size_t)net * 128, 128, biasbuf, 128, tid);
  wait_async(); __syncthreads();
  layer_small<2, 8, 4, 0>(hAbuf, wbuf, biasbuf, h1buf, wave, lane);   // -> L7 input
  __syncthreads();

  // ---------- layer 7: tanh + row-sum, streamed N ----------
  const _Float16* W7Fn = W7F + (size_t)net * 7 * 14336;
  const float* b7n = b7 + (size_t)net * 784;
  const v16h* As = (const v16h*)h1buf;
  for (int nc = 0; nc < 7; ++nc) {
    __syncthreads();                              // prev chunk reads done
    copy_lds(w7frag, W7Fn + (size_t)nc * 14336, 14336, tid);
    for (int i = tid; i < 112; i += NTHREADS) b7buf[i] = b7n[nc * 112 + i];
    wait_async(); __syncthreads();

    v16h a4[4];
#pragma unroll
    for (int kc = 0; kc < 4; ++kc) a4[kc] = As[(wave * 4 + kc) * 32 + lane];

    const v16h* Bs = (const v16h*)w7frag;
    const int ncol  = lane & 15;
    const int mbase = wave * 16 + ((lane & 16) ? 8 : 0);
#pragma unroll
    for (int nt = 0; nt < 7; ++nt) {
      v16h breg[4];
#pragma unroll
      for (int kc = 0; kc < 4; ++kc) breg[kc] = Bs[(nt * 4 + kc) * 32 + lane];
      SCHED_FENCE();
      v8f acc = {};
#pragma unroll
      for (int kc = 0; kc < 4; ++kc) acc = wmma16(a4[kc], breg[kc], acc);
      const float bv = b7buf[nt * 16 + ncol];
#pragma unroll
      for (int r = 0; r < 8; ++r) {
        float v = fast_tanh(acc[r] + bv);
        v += __shfl_xor(v, 1, 32);
        v += __shfl_xor(v, 2, 32);
        v += __shfl_xor(v, 4, 32);
        v += __shfl_xor(v, 8, 32);
        if ((lane & 15) == 0) atomicAdd(&rowsum[mbase + r], v);
      }
    }
  }
  __syncthreads();

  if (tid < BT) out[(size_t)(brow + tid) * NNETS + net] = rowsum[tid];
}

extern "C" void kernel_launch(void* const* d_in, const int* in_sizes, int n_in,
                              void* d_out, int out_size, void* d_ws, size_t ws_size,
                              hipStream_t stream) {
  (void)in_sizes; (void)n_in; (void)out_size; (void)ws_size;
  const float* x = (const float*)d_in[0];
  const float* W[8];
  const float* b[8];
  for (int l = 0; l < 8; ++l) {
    W[l] = (const float*)d_in[1 + 2 * l];
    b[l] = (const float*)d_in[2 + 2 * l];
  }
  _Float16* XF  = (_Float16*)d_ws;                 // 17.6 MB total in d_ws
  _Float16* W0F = XF  + XF_HALVES;
  _Float16* WSF = W0F + W0F_HALVES;
  _Float16* W7F = WSF + WSF_HALVES;

  prep_x_kernel<<<dim3(64, 25), 256, 0, stream>>>(x, XF);
  prep_w_kernel<<<dim3(38, 10), 256, 0, stream>>>(
      W[0], W[1], W[2], W[3], W[4], W[5], W[6], W[7], W0F, WSF, W7F);

  dim3 grid(BATCH / BT, NNETS);
  ae_fused_kernel<<<grid, NTHREADS, 0, stream>>>(
      XF, W0F, WSF, W7F,
      b[0], b[1], b[2], b[3], b[4], b[5], b[6], b[7],
      (float*)d_out);
}
